// DCGRUCell_29764123361433
// MI455X (gfx1250) — matrix operand order repaired and checked by reading, not compiled, not run on hardware
//
#include <hip/hip_runtime.h>
#include <hip/hip_bf16.h>
#include <cstdint>

// ---------------------------------------------------------------------------
// DCGRU cell forward for MI455X (gfx1250, wave32, WMMA).
// All heavy math runs through v_wmma_f32_16x16x32_f16 with f32 accumulation.
// ---------------------------------------------------------------------------

typedef __attribute__((ext_vector_type(16))) _Float16 v16h;
typedef __attribute__((ext_vector_type(8)))  float    v8f;

union F16Frag {
    v16h     h;
    uint4    u[2];
    _Float16 e[16];
};

#define NNODES 1024
#define BATCH  64
#define FDIM   128           // D_IN + UNITS
#define XCOLS  8192          // BATCH * FDIM
#define BN     65536         // BATCH * NNODES

// ---------------------------------------------------------------------------
// Diffusion GEMM: O[1024, 8192] = alpha * (L[1024,1024] @ X[1024,8192]) - Res
// Block tile 128x128, 8 waves (4 along M, 2 along N), wave tile 32x64,
// K-step 32 -> one v_wmma_f32_16x16x32_f16 per 16x16x32 sub-problem.
// ---------------------------------------------------------------------------
template<int HAS_RES>
__global__ void __launch_bounds__(256)
diff_gemm_kernel(const _Float16* __restrict__ L,
                 const _Float16* __restrict__ X,
                 const _Float16* __restrict__ Res,
                 _Float16* __restrict__ O,
                 float alpha)
{
    __shared__ __align__(16) _Float16 sA[128 * 32];   // [m][k]
    __shared__ __align__(16) _Float16 sB[128 * 32];   // transposed: [n][k]

    const int t    = threadIdx.x;
    const int lane = t & 31;
    const int wave = t >> 5;
    const int waveM = wave >> 1;       // 0..3
    const int waveN = wave & 1;        // 0..1
    const int rowBase = blockIdx.y * 128;
    const int colBase = blockIdx.x * 128;
    const int hb = lane >> 4;          // half-wave select
    const int ml = lane & 15;

    const v8f vzero = {0.f, 0.f, 0.f, 0.f, 0.f, 0.f, 0.f, 0.f};
    v8f acc[2][4];
#pragma unroll
    for (int i = 0; i < 2; ++i)
#pragma unroll
        for (int j = 0; j < 4; ++j) acc[i][j] = vzero;

    // staging work split: A = 128 rows x 32 halves, B = 32 rows x 128 halves
    const int stA_r = t >> 1, stA_c = (t & 1) * 16;
    const int stB_k = t >> 3, stB_c = (t & 7) * 16;

    for (int k0 = 0; k0 < 1024; k0 += 32) {
        // ---- stage A tile (row-major) ----
        {
            const uint4* src = reinterpret_cast<const uint4*>(
                L + (size_t)(rowBase + stA_r) * 1024 + k0 + stA_c);
            uint4* dst = reinterpret_cast<uint4*>(&sA[stA_r * 32 + stA_c]);
            dst[0] = src[0];
            dst[1] = src[1];
        }
        // ---- stage B tile transposed into [n][k] ----
        {
            const _Float16* src =
                X + (size_t)(k0 + stB_k) * XCOLS + colBase + stB_c;
            F16Frag tmp;
            tmp.u[0] = reinterpret_cast<const uint4*>(src)[0];
            tmp.u[1] = reinterpret_cast<const uint4*>(src)[1];
#pragma unroll
            for (int i = 0; i < 16; ++i)
                sB[(stB_c + i) * 32 + stB_k] = tmp.e[i];
        }
        if (k0 + 32 < 1024) {   // global_prefetch_b8 of next K-step
            __builtin_prefetch(L + (size_t)(rowBase + stA_r) * 1024 + k0 + 32 + stA_c, 0, 0);
            __builtin_prefetch(X + (size_t)(k0 + 32 + stB_k) * XCOLS + colBase + stB_c, 0, 0);
        }
        __syncthreads();

        // ---- build fragments (ISA VGPR layouts) ----
        // A 16x32 f16: lane l holds K runs {base..base+7, base+16..base+23},
        // base = (l>>4)*8, row M = l&15.
        F16Frag afr[2];
#pragma unroll
        for (int mt = 0; mt < 2; ++mt) {
            const uint4* p = reinterpret_cast<const uint4*>(
                &sA[(waveM * 32 + mt * 16 + ml) * 32 + hb * 8]);
            afr[mt].u[0] = p[0];
            afr[mt].u[1] = p[2];   // +16 halves
        }
        // B 32x16 f16: lane l holds column N = l&15, K runs {base..base+7, base+16..base+23}, base = (l>>4)*8.
        F16Frag bfr[4];
#pragma unroll
        for (int nt = 0; nt < 4; ++nt) {
            const uint4* p = reinterpret_cast<const uint4*>(
                &sB[(waveN * 64 + nt * 16 + ml) * 32 + hb * 8]);
            bfr[nt].u[0] = p[0];
            bfr[nt].u[1] = p[2];
        }
#pragma unroll
        for (int mt = 0; mt < 2; ++mt)
#pragma unroll
            for (int nt = 0; nt < 4; ++nt)
                acc[mt][nt] = __builtin_amdgcn_wmma_f32_16x16x32_f16(
                    false, afr[mt].h, false, bfr[nt].h,
                    (short)0, acc[mt][nt], false, false);
        __syncthreads();
    }

    // ---- epilogue: C/D layout -> M = vgpr + 8*(l>>4), N = l&15 ----
#pragma unroll
    for (int mt = 0; mt < 2; ++mt) {
#pragma unroll
        for (int nt = 0; nt < 4; ++nt) {
            const int gn = colBase + waveN * 64 + nt * 16 + ml;
#pragma unroll
            for (int vg = 0; vg < 8; ++vg) {
                const int gm = rowBase + waveM * 32 + mt * 16 + vg + 8 * hb;
                const size_t idx = (size_t)gm * XCOLS + gn;
                float v = alpha * acc[mt][nt][vg];
                if (HAS_RES) v -= (float)Res[idx];
                O[idx] = (_Float16)v;
            }
        }
    }
}

// ---------------------------------------------------------------------------
// Projection GEMM over the 5 diffusion stages (K = 5*128 = 640), with fused
// GRU gate epilogues.
//   MODE 1 (NC=128): value = sigmoid(xk@W1 + b1); write r*hx (f16) and u (f32)
//   MODE 2 (NC=64):  c = tanh(xk@W2 + b2); out = u*hx + (1-u)*c  -> d_out
// Wp is pre-packed stage-major: Wp[m*128+f, j] = W[f*5+m, j].
// ---------------------------------------------------------------------------
template<int NC, int MODE>
__global__ void __launch_bounds__(256)
proj_gemm_kernel(const _Float16* __restrict__ S0, const _Float16* __restrict__ S1,
                 const _Float16* __restrict__ S2, const _Float16* __restrict__ S3,
                 const _Float16* __restrict__ S4,
                 const _Float16* __restrict__ Wp, const float* __restrict__ bias,
                 const float* __restrict__ hx,   const float* __restrict__ ubuf,
                 _Float16* __restrict__ rh, float* __restrict__ uout,
                 float* __restrict__ dout)
{
    constexpr int MT = (NC == 128) ? 2 : 1;   // M tiles per wave
    __shared__ __align__(16) _Float16 sA[128 * 32];
    __shared__ __align__(16) _Float16 sB[NC * 32];    // transposed [n][k]

    const int t    = threadIdx.x;
    const int lane = t & 31;
    const int wave = t >> 5;
    const int waveM = (NC == 128) ? (wave >> 1) : wave;
    const int waveN = (NC == 128) ? (wave & 1) : 0;
    const int r0 = blockIdx.x * 128;
    const int hb = lane >> 4;
    const int ml = lane & 15;

    const _Float16* SP[5] = {S0, S1, S2, S3, S4};

    const v8f vzero = {0.f, 0.f, 0.f, 0.f, 0.f, 0.f, 0.f, 0.f};
    v8f acc[MT][4];
#pragma unroll
    for (int i = 0; i < MT; ++i)
#pragma unroll
        for (int j = 0; j < 4; ++j) acc[i][j] = vzero;

    // A staging: row r of this block = global sample row (b*1024 + n)
    const int ar = t >> 1, ac = (t & 1) * 16;
    const int gr = r0 + ar;
    const int bb = gr >> 10, nn = gr & 1023;
    const size_t aSrcBase = (size_t)nn * XCOLS + bb * FDIM;

    for (int ks = 0; ks < 20; ++ks) {           // 640 / 32
        const _Float16* Sm = SP[ks >> 2];       // 4 K-steps per stage
        const int f0 = (ks & 3) * 32;
        {
            const uint4* src =
                reinterpret_cast<const uint4*>(Sm + aSrcBase + f0 + ac);
            uint4* dst = reinterpret_cast<uint4*>(&sA[ar * 32 + ac]);
            dst[0] = src[0];
            dst[1] = src[1];
        }
        {
            const int k0 = ks * 32;
            if (NC == 128) {
                const int kk = t >> 3, j0 = (t & 7) * 16;
                const _Float16* src = Wp + (size_t)(k0 + kk) * NC + j0;
                F16Frag tmp;
                tmp.u[0] = reinterpret_cast<const uint4*>(src)[0];
                tmp.u[1] = reinterpret_cast<const uint4*>(src)[1];
#pragma unroll
                for (int i = 0; i < 16; ++i)
                    sB[(j0 + i) * 32 + kk] = tmp.e[i];
            } else {
                const int kk = t >> 3, j0 = (t & 7) * 8;
                const _Float16* src = Wp + (size_t)(k0 + kk) * NC + j0;
                F16Frag tmp;
                tmp.u[0] = reinterpret_cast<const uint4*>(src)[0];
#pragma unroll
                for (int i = 0; i < 8; ++i)
                    sB[(j0 + i) * 32 + kk] = tmp.e[i];
            }
        }
        __syncthreads();

        F16Frag afr[MT];
#pragma unroll
        for (int mt = 0; mt < MT; ++mt) {
            const uint4* p = reinterpret_cast<const uint4*>(
                &sA[(waveM * (MT * 16) + mt * 16 + ml) * 32 + hb * 8]);
            afr[mt].u[0] = p[0];
            afr[mt].u[1] = p[2];
        }
        F16Frag bfr[4];
#pragma unroll
        for (int nt = 0; nt < 4; ++nt) {
            const uint4* p = reinterpret_cast<const uint4*>(
                &sB[(waveN * 64 + nt * 16 + ml) * 32 + hb * 8]);
            bfr[nt].u[0] = p[0];
            bfr[nt].u[1] = p[2];
        }
#pragma unroll
        for (int mt = 0; mt < MT; ++mt)
#pragma unroll
            for (int nt = 0; nt < 4; ++nt)
                acc[mt][nt] = __builtin_amdgcn_wmma_f32_16x16x32_f16(
                    false, afr[mt].h, false, bfr[nt].h,
                    (short)0, acc[mt][nt], false, false);
        __syncthreads();
    }

    // ---- fused GRU gate epilogue ----
#pragma unroll
    for (int mt = 0; mt < MT; ++mt) {
#pragma unroll
        for (int nt = 0; nt < 4; ++nt) {
            const int gn = waveN * 64 + nt * 16 + ml;   // 0..NC-1
#pragma unroll
            for (int vg = 0; vg < 8; ++vg) {
                const int gm = r0 + waveM * (MT * 16) + mt * 16 + vg + 8 * hb;
                float v = acc[mt][nt][vg] + bias[gn];
                const size_t hidx = (size_t)gm * 64 + (gn & 63);
                if (MODE == 1) {
                    const float s = 1.0f / (1.0f + __expf(-v));
                    if (gn < 64)
                        rh[hidx] = (_Float16)(s * hx[hidx]);   // r * hx
                    else
                        uout[hidx] = s;                        // u
                } else {
                    const float c  = tanhf(v);
                    const float uu = ubuf[hidx];
                    dout[hidx] = uu * hx[hidx] + (1.0f - uu) * c;
                }
            }
        }
    }
}

// ---------------------------------------------------------------------------
// Prep / glue kernels
// ---------------------------------------------------------------------------
__global__ void cvt_f32_to_f16_kernel(const float* __restrict__ in,
                                      _Float16* __restrict__ out, int count)
{
    int i = blockIdx.x * blockDim.x + threadIdx.x;
    const int stride = gridDim.x * blockDim.x;
    for (; i < count; i += stride) out[i] = (_Float16)in[i];
}

// X0[n, b*128 + f] = f16(concat(inputs, hx)[b*1024+n, f])
__global__ void build_x0_kernel(const float* __restrict__ inp,
                                const float* __restrict__ hx,
                                _Float16* __restrict__ X0)
{
    const size_t base = ((size_t)blockIdx.x * blockDim.x + threadIdx.x) * 8;
    const int n   = (int)(base >> 13);
    const int rem = (int)(base & 8191);
    const int b = rem >> 7, f = rem & 127;
    const float* src = (f < 64)
        ? (inp + ((size_t)b * NNODES + n) * 64 + f)
        : (hx  + ((size_t)b * NNODES + n) * 64 + (f - 64));
    __align__(16) _Float16 o[8];
#pragma unroll
    for (int j = 0; j < 8; ++j) o[j] = (_Float16)src[j];
    *reinterpret_cast<uint4*>(X0 + base) = *reinterpret_cast<const uint4*>(o);
}

// Wp[m*128+f, j] = f16(W[f*5+m, j])  (stage-major repack of projection weight)
__global__ void pack_w_kernel(const float* __restrict__ W,
                              _Float16* __restrict__ Wp, int NC, int total)
{
    int i = blockIdx.x * blockDim.x + threadIdx.x;
    const int stride = gridDim.x * blockDim.x;
    for (; i < total; i += stride) {
        const int k = i / NC, j = i - k * NC;
        const int m = k >> 7, f = k & 127;
        Wp[i] = (_Float16)W[(size_t)(f * 5 + m) * NC + j];
    }
}

// X0 hx-half <- r*hx (done after proj1 completes to avoid WAR race)
__global__ void scatter_rh_kernel(const _Float16* __restrict__ rh,
                                  _Float16* __restrict__ X0)
{
    const size_t base = ((size_t)blockIdx.x * blockDim.x + threadIdx.x) * 8;
    const int r = (int)(base >> 6);
    const int j = (int)(base & 63);
    const int b = r >> 10, n = r & 1023;
    *reinterpret_cast<uint4*>(X0 + (size_t)n * XCOLS + b * FDIM + 64 + j) =
        *reinterpret_cast<const uint4*>(rh + base);
}

// ---------------------------------------------------------------------------
extern "C" void kernel_launch(void* const* d_in, const int* in_sizes, int n_in,
                              void* d_out, int out_size, void* d_ws, size_t ws_size,
                              hipStream_t stream)
{
    const float* inputs = (const float*)d_in[0];
    const float* hx     = (const float*)d_in[1];
    const float* L0     = (const float*)d_in[2];
    const float* L1     = (const float*)d_in[3];
    const float* W1     = (const float*)d_in[4];
    const float* b1     = (const float*)d_in[5];
    const float* W2     = (const float*)d_in[6];
    const float* b2     = (const float*)d_in[7];
    float* dout = (float*)d_out;

    char* ws = (char*)d_ws;
    const size_t SZ_L = (size_t)NNODES * NNODES * 2;   // 2 MB  f16 Laplacian
    const size_t SZ_S = (size_t)NNODES * XCOLS * 2;    // 16 MB f16 stage

    _Float16* L0h = (_Float16*)(ws);
    _Float16* L1h = (_Float16*)(ws + SZ_L);
    _Float16* X0  = (_Float16*)(ws + 2 * SZ_L);
    _Float16* Y1a = (_Float16*)(ws + 2 * SZ_L + 1 * SZ_S);
    _Float16* Y1b = (_Float16*)(ws + 2 * SZ_L + 2 * SZ_S);
    _Float16* Y2a = (_Float16*)(ws + 2 * SZ_L + 3 * SZ_S);
    _Float16* Y2b = (_Float16*)(ws + 2 * SZ_L + 4 * SZ_S);
    _Float16* Wp1 = (_Float16*)(ws + 2 * SZ_L + 5 * SZ_S);          // 160 KB
    _Float16* Wp2 = Wp1 + 640 * 128;                                 // 80 KB
    float*    ubuf = (float*)(ws + 2 * SZ_L + 5 * SZ_S + (256 << 10));
    _Float16* rh   = (_Float16*)((char*)ubuf + (size_t)BN * 64 * 4);

    // ---- prep: f16 conversions + weight repack ----
    cvt_f32_to_f16_kernel<<<1024, 256, 0, stream>>>(L0, L0h, NNODES * NNODES);
    cvt_f32_to_f16_kernel<<<1024, 256, 0, stream>>>(L1, L1h, NNODES * NNODES);
    pack_w_kernel<<<320, 256, 0, stream>>>(W1, Wp1, 128, 640 * 128);
    pack_w_kernel<<<160, 256, 0, stream>>>(W2, Wp2, 64, 640 * 64);
    build_x0_kernel<<<4096, 256, 0, stream>>>(inputs, hx, X0);

    const dim3 dgrid(XCOLS / 128, NNODES / 128);   // 64 x 8

    // ---- gconv 1: diffusion + gated projection ----
    diff_gemm_kernel<0><<<dgrid, 256, 0, stream>>>(L0h, X0,  nullptr, Y1a, 1.0f);
    diff_gemm_kernel<0><<<dgrid, 256, 0, stream>>>(L1h, X0,  nullptr, Y1b, 1.0f);
    diff_gemm_kernel<1><<<dgrid, 256, 0, stream>>>(L0h, Y1a, X0,      Y2a, 2.0f);
    diff_gemm_kernel<1><<<dgrid, 256, 0, stream>>>(L1h, Y1b, X0,      Y2b, 2.0f);
    proj_gemm_kernel<128, 1><<<BN / 128, 256, 0, stream>>>(
        X0, Y1a, Y1b, Y2a, Y2b, Wp1, b1, hx, nullptr, rh, ubuf, nullptr);
    scatter_rh_kernel<<<2048, 256, 0, stream>>>(rh, X0);

    // ---- gconv 2: diffusion + candidate + GRU blend ----
    diff_gemm_kernel<0><<<dgrid, 256, 0, stream>>>(L0h, X0,  nullptr, Y1a, 1.0f);
    diff_gemm_kernel<0><<<dgrid, 256, 0, stream>>>(L1h, X0,  nullptr, Y1b, 1.0f);
    diff_gemm_kernel<1><<<dgrid, 256, 0, stream>>>(L0h, Y1a, X0,      Y2a, 2.0f);
    diff_gemm_kernel<1><<<dgrid, 256, 0, stream>>>(L1h, Y1b, X0,      Y2b, 2.0f);
    proj_gemm_kernel<64, 2><<<BN / 128, 256, 0, stream>>>(
        X0, Y1a, Y1b, Y2a, Y2b, Wp2, b2, hx, ubuf, nullptr, nullptr, dout);
}
